// GAT_31250182045886
// MI455X (gfx1250) — matrix operand order repaired and checked by reading, not compiled, not run on hardware
//
#include <hip/hip_runtime.h>
#include <hip/hip_bf16.h>

// ---------------------------------------------------------------------------
// GAT 2-layer forward for MI455X (gfx1250, wave32).
// GEMMs via v_wmma_f32_16x16x32_bf16; edge softmax/aggregation via native
// global f32/u32 atomics (L2-resident working set).
// ---------------------------------------------------------------------------

typedef __attribute__((ext_vector_type(16))) __bf16 v16bf;
typedef __attribute__((ext_vector_type(8)))  float  v8f;

#define NN 50000
#define NE 1600000

// float -> bf16, round-to-nearest-even
__device__ __forceinline__ unsigned short f2bf(float f) {
  unsigned u = __float_as_uint(f);
  u += 0x7FFFu + ((u >> 16) & 1u);
  return (unsigned short)(u >> 16);
}
// order-preserving float<->uint transform (for atomicMax-based segment max)
__device__ __forceinline__ unsigned enc_ord(float f) {
  unsigned u = __float_as_uint(f);
  return (u & 0x80000000u) ? ~u : (u | 0x80000000u);
}
__device__ __forceinline__ float dec_ord(unsigned u) {
  u = (u & 0x80000000u) ? (u & 0x7FFFFFFFu) : ~u;
  return __uint_as_float(u);
}

// -------------------------------- converts --------------------------------
__global__ void cvt_bf16_k(const float* __restrict__ x,
                           unsigned short* __restrict__ y, int n) {
  int i = blockIdx.x * blockDim.x + threadIdx.x;
  if (i < n) y[i] = f2bf(x[i]);
}

// W: [K, Nout] row-major  ->  Wt: [Nout, 128] bf16 (K is always 128 here)
__global__ void cvt_wt_k(const float* __restrict__ W,
                         unsigned short* __restrict__ Wt, int K, int Nout) {
  int i = blockIdx.x * blockDim.x + threadIdx.x;
  if (i >= K * Nout) return;
  int k = i / Nout, n = i - k * Nout;
  Wt[n * 128 + k] = f2bf(W[i]);
}

// ------------------------------ WMMA GEMM ---------------------------------
// C[M,Nout] = A[M,128](bf16) * B[128,Nout], B given transposed as Bt[Nout,128].
// One wave computes one 16x16 tile with 4x v_wmma_f32_16x16x32_bf16.
__global__ void __launch_bounds__(256)
gemm_bf16_wmma(const unsigned short* __restrict__ A,
               const unsigned short* __restrict__ Bt,
               float* __restrict__ C, int M, int Nout) {
  const int wave  = blockIdx.x * (blockDim.x >> 5) + (threadIdx.x >> 5);
  const int lane  = threadIdx.x & 31;
  const int ntile = Nout >> 4;
  const int total = (M >> 4) * ntile;
  if (wave >= total) return;               // wave-uniform: EXEC stays all-1s
  const int tm   = wave / ntile;
  const int tn   = wave - tm * ntile;
  const int half = lane >> 4;              // 0: lanes 0-15, 1: lanes 16-31
  const int l16  = lane & 15;

  // A lane layout (16-bit A 16x32): lanes 0-15 K={0..7,16..23}, lanes 16-31 +8
  const unsigned short* aptr = A  + (size_t)(tm * 16 + l16) * 128 + half * 8;
  // B lane layout (16-bit B 32x16): lanes 0-15 K={0..7,16..23}, lanes 16-31 +8
  const unsigned short* bptr = Bt + (size_t)(tn * 16 + l16) * 128 + half * 8;

  v8f acc = {};
#pragma unroll
  for (int ks = 0; ks < 4; ++ks) {
    union { uint4 q[2]; v16bf v; } a, b;
    a.q[0] = *(const uint4*)(aptr + ks * 32);
    a.q[1] = *(const uint4*)(aptr + ks * 32 + 16);
    b.q[0] = *(const uint4*)(bptr + ks * 32);
    b.q[1] = *(const uint4*)(bptr + ks * 32 + 16);
    acc = __builtin_amdgcn_wmma_f32_16x16x32_bf16(
        false, a.v, false, b.v, (short)0, acc, false, false);
  }
  // C/D layout: lane = column, VGPR r = row r (lanes 0-15) / row r+8 (16-31)
  const int crow0 = tm * 16 + half * 8;
  const int ccol  = tn * 16 + l16;
#pragma unroll
  for (int r = 0; r < 8; ++r)
    C[(size_t)(crow0 + r) * Nout + ccol] = acc[r];
}

// --------------------------- attention coeffs -----------------------------
// el[n,h] = dot(z[n,h,:], al[h,:]); er likewise.  D = 32 fixed.
template <int H>
__global__ void attn_coef_k(const float* __restrict__ z,
                            const float* __restrict__ al,
                            const float* __restrict__ ar,
                            float* __restrict__ el, float* __restrict__ er,
                            int NH) {
  int i = blockIdx.x * blockDim.x + threadIdx.x;
  if (i >= NH) return;
  int n = i / H, h = i - n * H;
  const float* zp = z + (size_t)n * (H * 32) + h * 32;
  float sl = 0.f, sr = 0.f;
#pragma unroll 8
  for (int d = 0; d < 32; ++d) {
    float zv = zp[d];
    sl += zv * al[h * 32 + d];
    sr += zv * ar[h * 32 + d];
  }
  el[i] = sl;
  er[i] = sr;
}

// ------------------------------- inits ------------------------------------
__global__ void init_l1_k(float* __restrict__ agg, unsigned* __restrict__ m,
                          float* __restrict__ s, int NH, int NHD) {
  int i = blockIdx.x * blockDim.x + threadIdx.x;
  if (i < NHD) agg[i] = 0.f;
  if (i < NH) { m[i] = 0u; s[i] = 0.f; }
}
__global__ void init_l2_k(float* __restrict__ y, const float* __restrict__ b2,
                          unsigned* __restrict__ m, float* __restrict__ s,
                          int N32) {
  int i = blockIdx.x * blockDim.x + threadIdx.x;
  if (i < N32) y[i] = b2[i & 31];
  if (i < NN) { m[i] = 0u; s[i] = 0.f; }
}

// ------------------------- edge softmax passes ----------------------------
template <int H>
__global__ void edge_max_k(const int* __restrict__ src, const int* __restrict__ dst,
                           const float* __restrict__ el, const float* __restrict__ er,
                           unsigned* __restrict__ m, int EH) {
  int i = blockIdx.x * blockDim.x + threadIdx.x;
  if (i >= EH) return;
  int e = i / H, h = i - e * H;
  int sn = src[e], dn = dst[e];
  float v = el[sn * H + h] + er[dn * H + h];
  v = v > 0.f ? v : 0.2f * v;                     // leaky_relu(0.2)
  atomicMax(m + dn * H + h, enc_ord(v));
}

template <int H>
__global__ void edge_sum_k(const int* __restrict__ src, const int* __restrict__ dst,
                           const float* __restrict__ el, const float* __restrict__ er,
                           const unsigned* __restrict__ m, float* __restrict__ s,
                           int EH) {
  int i = blockIdx.x * blockDim.x + threadIdx.x;
  if (i >= EH) return;
  int e = i / H, h = i - e * H;
  int sn = src[e], dn = dst[e];
  float v = el[sn * H + h] + er[dn * H + h];
  v = v > 0.f ? v : 0.2f * v;
  atomicAdd(s + dn * H + h, __expf(v - dec_ord(m[dn * H + h])));
}

// layer1 messages: H=4, D=32 -> 128 floats/edge, one wave/edge, float4/lane
__global__ void edge_msg1_k(const int* __restrict__ src, const int* __restrict__ dst,
                            const float* __restrict__ el, const float* __restrict__ er,
                            const unsigned* __restrict__ m, const float* __restrict__ s,
                            const float* __restrict__ z, float* __restrict__ agg) {
  int wid  = (blockIdx.x * blockDim.x + threadIdx.x) >> 5;
  int lane = threadIdx.x & 31;
  if (wid >= NE) return;
  int sn = src[wid], dn = dst[wid];
  int h = lane >> 3;                               // floats [4*lane..4*lane+3] -> head
  float e = el[sn * 4 + h] + er[dn * 4 + h];
  e = e > 0.f ? e : 0.2f * e;
  float alpha = __expf(e - dec_ord(m[dn * 4 + h])) / s[dn * 4 + h];
  const float4 zv = *(const float4*)(z + (size_t)sn * 128 + lane * 4);
  float* op = agg + (size_t)dn * 128 + lane * 4;
  atomicAdd(op + 0, alpha * zv.x);
  atomicAdd(op + 1, alpha * zv.y);
  atomicAdd(op + 2, alpha * zv.z);
  atomicAdd(op + 3, alpha * zv.w);
}

// layer2 messages: H=1, D=32 -> 32 floats/edge, one wave/edge, 1 float/lane
__global__ void edge_msg2_k(const int* __restrict__ src, const int* __restrict__ dst,
                            const float* __restrict__ el, const float* __restrict__ er,
                            const unsigned* __restrict__ m, const float* __restrict__ s,
                            const float* __restrict__ z, float* __restrict__ y) {
  int wid  = (blockIdx.x * blockDim.x + threadIdx.x) >> 5;
  int lane = threadIdx.x & 31;
  if (wid >= NE) return;
  int sn = src[wid], dn = dst[wid];
  float e = el[sn] + er[dn];
  e = e > 0.f ? e : 0.2f * e;
  float alpha = __expf(e - dec_ord(m[dn])) / s[dn];
  float zv = z[(size_t)sn * 32 + lane];
  atomicAdd(y + (size_t)dn * 32 + lane, alpha * zv);
}

// x = elu(agg + b1), stored directly as bf16 for layer-2 GEMM A operand
__global__ void act_cvt_k(const float* __restrict__ agg, const float* __restrict__ b1,
                          unsigned short* __restrict__ xb, int n) {
  int i = blockIdx.x * blockDim.x + threadIdx.x;
  if (i >= n) return;
  float v = agg[i] + b1[i & 127];
  v = v > 0.f ? v : (__expf(v) - 1.f);             // elu
  xb[i] = f2bf(v);
}

// ---------------------------------------------------------------------------
extern "C" void kernel_launch(void* const* d_in, const int* in_sizes, int n_in,
                              void* d_out, int out_size, void* d_ws, size_t ws_size,
                              hipStream_t stream) {
  const float* h   = (const float*)d_in[0];
  const int*   src = (const int*)d_in[1];
  const int*   dst = (const int*)d_in[2];
  const float* W1  = (const float*)d_in[3];
  const float* al1 = (const float*)d_in[4];
  const float* ar1 = (const float*)d_in[5];
  const float* b1  = (const float*)d_in[6];
  const float* W2  = (const float*)d_in[7];
  const float* al2 = (const float*)d_in[8];
  const float* ar2 = (const float*)d_in[9];
  const float* b2  = (const float*)d_in[10];
  float* y = (float*)d_out;

  // workspace layout (all 256B-aligned by construction, ~67.2 MB total)
  char* p = (char*)d_ws;
  unsigned short* hb = (unsigned short*)p; p += (size_t)NN * 128 * 2;  // A operand (reused for x in layer 2)
  unsigned short* Wt = (unsigned short*)p; p += 128 * 128 * 2;         // transposed bf16 weights (reused)
  float*    z    = (float*)p;              p += (size_t)NN * 128 * 4;  // projections (reused for z2)
  float*    el   = (float*)p;              p += NN * 4 * 4;
  float*    er   = (float*)p;              p += NN * 4 * 4;
  unsigned* mbuf = (unsigned*)p;           p += NN * 4 * 4;
  float*    sbuf = (float*)p;              p += NN * 4 * 4;
  float*    agg  = (float*)p;              p += (size_t)NN * 128 * 4;  // layer-1 aggregation

  auto cdiv = [](int a, int b) { return (a + b - 1) / b; };
  const int B = 256;

  // ---------------- layer 1 (H=4, D=32) ----------------
  cvt_bf16_k<<<cdiv(NN * 128, B), B, 0, stream>>>(h, hb, NN * 128);
  cvt_wt_k<<<cdiv(128 * 128, B), B, 0, stream>>>(W1, Wt, 128, 128);
  gemm_bf16_wmma<<<cdiv((NN / 16) * (128 / 16), 8), B, 0, stream>>>(hb, Wt, z, NN, 128);
  attn_coef_k<4><<<cdiv(NN * 4, B), B, 0, stream>>>(z, al1, ar1, el, er, NN * 4);
  init_l1_k<<<cdiv(NN * 128, B), B, 0, stream>>>(agg, mbuf, sbuf, NN * 4, NN * 128);
  edge_max_k<4><<<cdiv(NE * 4, B), B, 0, stream>>>(src, dst, el, er, mbuf, NE * 4);
  edge_sum_k<4><<<cdiv(NE * 4, B), B, 0, stream>>>(src, dst, el, er, mbuf, sbuf, NE * 4);
  edge_msg1_k<<<cdiv(NE * 32, B), B, 0, stream>>>(src, dst, el, er, mbuf, sbuf, z, agg);
  act_cvt_k<<<cdiv(NN * 128, B), B, 0, stream>>>(agg, b1, hb, NN * 128);

  // ---------------- layer 2 (H=1, D=32) ----------------
  cvt_wt_k<<<cdiv(128 * 32, B), B, 0, stream>>>(W2, Wt, 128, 32);
  gemm_bf16_wmma<<<cdiv((NN / 16) * (32 / 16), 8), B, 0, stream>>>(hb, Wt, z, NN, 32);
  attn_coef_k<1><<<cdiv(NN, B), B, 0, stream>>>(z, al2, ar2, el, er, NN);
  init_l2_k<<<cdiv(NN * 32, B), B, 0, stream>>>(y, b2, mbuf, sbuf, NN * 32);
  edge_max_k<1><<<cdiv(NE, B), B, 0, stream>>>(src, dst, el, er, mbuf, NE);
  edge_sum_k<1><<<cdiv(NE, B), B, 0, stream>>>(src, dst, el, er, mbuf, sbuf, NE);
  edge_msg2_k<<<cdiv(NE * 32, B), B, 0, stream>>>(src, dst, el, er, mbuf, sbuf, z, y);
}
